// LSTM_62053687493021
// MI455X (gfx1250) — matrix operand order repaired and checked by reading, not compiled, not run on hardware
//
#include <hip/hip_runtime.h>
#include <cstdint>
#include <cstddef>

typedef __attribute__((ext_vector_type(16))) _Float16 v16h;
typedef __attribute__((ext_vector_type(8)))  float    v8f;

#define HDIM    128
#define G4      512
#define TSTEPS  1024
#define BATCH   128
#define BTILE   16
#define THREADS 512

// d_out layout: out [B*T] | hN [3*B*H] | cN [3*B*H]
#define OUT_H_OFF 131072
#define OUT_C_OFF 180224

__device__ __forceinline__ float fast_tanh(float x) {
#if __has_builtin(__builtin_amdgcn_tanhf)
    return __builtin_amdgcn_tanhf(x);
#elif __has_builtin(__builtin_amdgcn_tanh_f32)
    return __builtin_amdgcn_tanh_f32(x);
#else
    return tanhf(x);
#endif
}
__device__ __forceinline__ float sigf(float x) {
    // sigmoid(x) = 0.5*tanh(0.5x) + 0.5  (single hardware trans op on gfx1250)
    return __builtin_fmaf(0.5f, fast_tanh(0.5f * x), 0.5f);
}

// WGP-scope prefetch: pulls into ALL cache levels (ISA 10.5, scope 0),
// unlike __builtin_prefetch which emits SCOPE_SE (GL2 only).
__device__ __forceinline__ void prefetch_wgp(const void* p) {
    asm volatile("global_prefetch_b8 %0, off" :: "v"(p) : "memory");
}

union FragU {
    v16h     v;
    uint32_t u[8];
    _Float16 h[16];
};

// One workgroup = 16 batch rows, scans all T steps of one layer.
// 16 waves; wave w owns gate columns [32w, 32w+32) as two 16-wide N-tiles.
// Weights held as WMMA B-fragments in VGPRs for the whole scan.
template <int DIN, bool XF32>
__global__ __launch_bounds__(THREADS)
void lstm_layer_kernel(const void*  __restrict__ xin,   // [B,T,DIN] f32 (XF32) or f16
                       const float* __restrict__ h0l,   // [B,H]
                       const float* __restrict__ c0l,   // [B,H]
                       const float* __restrict__ Wih,   // [4H,DIN] f32
                       const float* __restrict__ Whh,   // [4H,H]   f32
                       const float* __restrict__ bih,   // [4H]
                       const float* __restrict__ bhh,   // [4H]
                       _Float16*    __restrict__ yout,  // [B,T,H] f16
                       float*       __restrict__ dout,
                       int layer)
{
    constexpr int KX = DIN / 32;

    // Gate matrix, gate-interleaved: Gs[m][j][{i,f,g,o}]  (32 KB)
    __shared__ float    Gs[BTILE * G4];
    __shared__ _Float16 hs[BTILE * HDIM];   // hidden state, f16 (4 KB)

    const int tid   = threadIdx.x;
    const int lane  = tid & 31;
    const int wave  = tid >> 5;
    const int lm    = lane & 15;            // M-row (A/C) or N-col (B/C)
    const int lg    = lane >> 4;            // lane half
    const int brow0 = blockIdx.x * BTILE;
    const int n0    = wave * 32;

    // Interleaved gate columns for this lane's two N-tiles
    const int nA   = n0 + lm;
    const int nB   = nA + 16;
    const int colA = ((nA & 127) << 2) | (nA >> 7);
    const int colB = ((nB & 127) << 2) | (nB >> 7);

    // ---- Build register-resident B fragments (B layout: 32x16, K = kb*32 + 16*lg + 2v (+1)) ----
    FragU Bh[2][4];
    FragU Bx[2][KX];
    float biasv[2];
#pragma unroll
    for (int tt = 0; tt < 2; ++tt) {
        const int n = n0 + tt * 16 + lm;
        biasv[tt]   = bih[n] + bhh[n];
#pragma unroll
        for (int kb = 0; kb < 4; ++kb)
#pragma unroll
            for (int v = 0; v < 8; ++v) {
                const int k             = kb * 32 + lg * 16 + 2 * v;
                Bh[tt][kb].h[2 * v]     = (_Float16)Whh[n * HDIM + k];
                Bh[tt][kb].h[2 * v + 1] = (_Float16)Whh[n * HDIM + k + 1];
            }
#pragma unroll
        for (int kb = 0; kb < KX; ++kb)
#pragma unroll
            for (int v = 0; v < 8; ++v) {
                const int k             = kb * 32 + lg * 16 + 2 * v;
                Bx[tt][kb].h[2 * v]     = (_Float16)Wih[n * DIN + k];
                Bx[tt][kb].h[2 * v + 1] = (_Float16)Wih[n * DIN + k + 1];
            }
    }

    // A layout (16x32 f16): lane half g, VGPR v holds K = kb*32 + avOff(v,g) (+1)
    int avOff[8];
#pragma unroll
    for (int v = 0; v < 8; ++v)
        avOff[v] = 2 * v + 16 * lg;

    // ---- init h (LDS, f16) and c (registers, fp32, 4 cells per thread) ----
    float creg[4];
#pragma unroll
    for (int u = 0; u < 4; ++u) {
        const int e = tid + u * THREADS;
        const int m = e >> 7, j = e & 127;
        creg[u]          = c0l[(brow0 + m) * HDIM + j];
        hs[m * HDIM + j] = (_Float16)h0l[(brow0 + m) * HDIM + j];
    }
    __syncthreads();

    const float*    xf32 = (const float*)xin;
    const _Float16* xf16 = (const _Float16*)xin;
    const size_t    xbase = (size_t)(brow0 + lm) * TSTEPS;   // row for this lane's A loads

#pragma unroll 1
    for (int t = 0; t < TSTEPS; ++t) {
        // ---- issue x_t loads FIRST (latency hidden behind h phase) ----
        FragU ax[KX];
        const size_t xrow = (xbase + (size_t)t) * DIN;
#pragma unroll
        for (int kb = 0; kb < KX; ++kb)
#pragma unroll
            for (int v = 0; v < 8; ++v) {
                const size_t gb = xrow + (size_t)(kb * 32 + avOff[v]);
                if (XF32) {
                    ax[kb].h[2 * v]     = (_Float16)xf32[gb];
                    ax[kb].h[2 * v + 1] = (_Float16)xf32[gb + 1];
                } else {
                    ax[kb].u[v] = *(const uint32_t*)(xf16 + gb);
                }
            }
        // WGP-scope prefetch of next step's x row (256B per lane = 2 cachelines)
        {
            const int tp = (t + 1 < TSTEPS) ? (t + 1) : t;
            const char* pr = XF32 ? (const char*)(xf32 + (xbase + (size_t)tp) * DIN)
                                  : (const char*)(xf16 + (xbase + (size_t)tp) * DIN);
            prefetch_wgp(pr);
            prefetch_wgp(pr + 128);
        }

        // ---- h fragments from LDS + WMMA chain ----
        FragU ah[4];
#pragma unroll
        for (int kb = 0; kb < 4; ++kb)
#pragma unroll
            for (int v = 0; v < 8; ++v)
                ah[kb].u[v] = *(const uint32_t*)(hs + (lm * HDIM + kb * 32 + avOff[v]));

        v8f acc0, acc1;
#pragma unroll
        for (int i = 0; i < 8; ++i) { acc0[i] = biasv[0]; acc1[i] = biasv[1]; }

        // h contribution first (LDS data), x contribution last (global data)
#pragma unroll
        for (int kb = 0; kb < 4; ++kb) {
            acc0 = __builtin_amdgcn_wmma_f32_16x16x32_f16(false, ah[kb].v, false, Bh[0][kb].v,
                                                          (short)0, acc0, false, false);
            acc1 = __builtin_amdgcn_wmma_f32_16x16x32_f16(false, ah[kb].v, false, Bh[1][kb].v,
                                                          (short)0, acc1, false, false);
        }
#pragma unroll
        for (int kb = 0; kb < KX; ++kb) {
            acc0 = __builtin_amdgcn_wmma_f32_16x16x32_f16(false, ax[kb].v, false, Bx[0][kb].v,
                                                          (short)0, acc0, false, false);
            acc1 = __builtin_amdgcn_wmma_f32_16x16x32_f16(false, ax[kb].v, false, Bx[1][kb].v,
                                                          (short)0, acc1, false, false);
        }

        // C/D layout: lane, VGPR v -> (M = v + 8*lg, N = lm); store gate-interleaved
#pragma unroll
        for (int v = 0; v < 8; ++v) {
            Gs[(v + 8 * lg) * G4 + colA] = acc0[v];
            Gs[(v + 8 * lg) * G4 + colB] = acc1[v];
        }
        __syncthreads();

        // ---- elementwise LSTM cell update: one b128 LDS read per cell ----
#pragma unroll
        for (int u = 0; u < 4; ++u) {
            const int e = tid + u * THREADS;
            const int m = e >> 7, j = e & 127;
            const float4 g4 = *(const float4*)&Gs[m * G4 + 4 * j];   // {i,f,g,o}
            const float c  = sigf(g4.y) * creg[u] + sigf(g4.x) * fast_tanh(g4.z);
            const float h  = sigf(g4.w) * fast_tanh(c);
            creg[u]        = c;
            hs[m * HDIM + j] = (_Float16)h;
            yout[((size_t)(brow0 + m) * TSTEPS + t) * HDIM + j] = (_Float16)h;
            if (t == TSTEPS - 1) {
                dout[OUT_H_OFF + layer * BATCH * HDIM + (brow0 + m) * HDIM + j] = h;
                dout[OUT_C_OFF + layer * BATCH * HDIM + (brow0 + m) * HDIM + j] = c;
            }
        }
        __syncthreads();
    }
}

// out[b*T+t] = relu(dot(y2[b,t,:], Wout) + bout)
__global__ __launch_bounds__(256)
void proj_kernel(const _Float16* __restrict__ y,
                 const float* __restrict__ Wout,
                 const float* __restrict__ bout,
                 float* __restrict__ out)
{
    const int idx = blockIdx.x * blockDim.x + threadIdx.x;   // 0 .. B*T-1
    const _Float16* yr = y + (size_t)idx * HDIM;
    float s = bout[0];
#pragma unroll 8
    for (int j = 0; j < HDIM; ++j) s += (float)yr[j] * Wout[j];
    out[idx] = s > 0.f ? s : 0.f;
}

extern "C" void kernel_launch(void* const* d_in, const int* in_sizes, int n_in,
                              void* d_out, int out_size, void* d_ws, size_t ws_size,
                              hipStream_t stream)
{
    (void)in_sizes; (void)n_in; (void)out_size; (void)ws_size;

    const float* x    = (const float*)d_in[0];
    const float* h0   = (const float*)d_in[1];   // [3,B,H]
    const float* c0   = (const float*)d_in[2];
    const float* Wih0 = (const float*)d_in[3];
    const float* Whh0 = (const float*)d_in[4];
    const float* bih0 = (const float*)d_in[5];
    const float* bhh0 = (const float*)d_in[6];
    const float* Wih1 = (const float*)d_in[7];
    const float* Whh1 = (const float*)d_in[8];
    const float* bih1 = (const float*)d_in[9];
    const float* bhh1 = (const float*)d_in[10];
    const float* Wih2 = (const float*)d_in[11];
    const float* Whh2 = (const float*)d_in[12];
    const float* bih2 = (const float*)d_in[13];
    const float* bhh2 = (const float*)d_in[14];
    const float* Wout = (const float*)d_in[15];
    const float* bout = (const float*)d_in[16];

    float*    out = (float*)d_out;
    _Float16* yb  = (_Float16*)d_ws;             // [B,T,H] f16, 32 MB

    const int BH = BATCH * HDIM;                 // 16384

    lstm_layer_kernel<64, true><<<BATCH / BTILE, THREADS, 0, stream>>>(
        x, h0, c0, Wih0, Whh0, bih0, bhh0, yb, out, 0);
    lstm_layer_kernel<128, false><<<BATCH / BTILE, THREADS, 0, stream>>>(
        yb, h0 + BH, c0 + BH, Wih1, Whh1, bih1, bhh1, yb, out, 1);
    lstm_layer_kernel<128, false><<<BATCH / BTILE, THREADS, 0, stream>>>(
        yb, h0 + 2 * BH, c0 + 2 * BH, Wih2, Whh2, bih2, bhh2, yb, out, 2);

    proj_kernel<<<(BATCH * TSTEPS) / 256, 256, 0, stream>>>(yb, Wout, bout, out);
}
